// ScaledDotProductAttention_19464791785931
// MI455X (gfx1250) — matrix operand order repaired and checked by reading, not compiled, not run on hardware
//
#include <hip/hip_runtime.h>
#include <hip/hip_bf16.h>

typedef __attribute__((ext_vector_type(16))) __bf16 v16bf;
typedef __attribute__((ext_vector_type(8)))  float  v8f;

#define SQd 2048
#define SKd 2048
#define DKd 128
#define DVd 128
#define KCHUNK 64
#define LOG2E 1.4426950408889634f

// LDS strides in bf16 elements; padded so 16 consecutive rows hit distinct banks
// while keeping 16B alignment for ds_load_b128 fragment reads.
#define KLDS_STRIDE 136   // 128 + 8 pad  (272 B per row), 64 rows (keys)
#define VLDS_STRIDE 72    // 64  + 8 pad  (144 B per row), 128 rows (dv)

union AFrag { v16bf v; uint4 q[2]; unsigned u[8]; unsigned short h[16]; };

__device__ __forceinline__ unsigned short f2bf(float x) {
  union { __bf16 h; unsigned short s; } u; u.h = (__bf16)x; return u.s;
}
__device__ __forceinline__ unsigned pk2(float a, float b) {
  union { __bf16 h[2]; unsigned u; } un;
  un.h[0] = (__bf16)a; un.h[1] = (__bf16)b; return un.u;
}
// swap data between lane L and lane L^16 (ds_swizzle group-of-32, xor=0x10)
__device__ __forceinline__ float swz16f(float x) {
  return __int_as_float(__builtin_amdgcn_ds_swizzle(__float_as_int(x), 0x401f));
}
__device__ __forceinline__ unsigned swz16u(unsigned x) {
  return (unsigned)__builtin_amdgcn_ds_swizzle((int)x, 0x401f);
}

extern "C" __global__ __launch_bounds__(256, 1)
void fa_fwd_bf16_wmma(const float* __restrict__ Q, const float* __restrict__ K,
                      const float* __restrict__ V, float* __restrict__ O) {
  __shared__ unsigned short lds_k[KCHUNK * KLDS_STRIDE];   // K chunk, row-major (key, dk), bf16
  __shared__ unsigned short lds_vt[DVd * VLDS_STRIDE];     // V chunk transposed (dv, key), bf16

  const int tid  = threadIdx.x;
  const int wave = tid >> 5;
  const int lane = tid & 31;
  const int g    = lane >> 4;   // half-wave group
  const int n    = lane & 15;   // column index (query for B/C/D, row for A frags)

  const int qt = blockIdx.x & 15;   // 16 query tiles of 128 per (b,h)
  const int bh = blockIdx.x >> 4;   // 64 (b,h) pairs
  const int qbase = qt * 128 + wave * 16;

  const size_t qrow0 = (size_t)bh * SQd;
  const size_t krow0 = (size_t)bh * SKd;

  // fold 1/sqrt(dk) and log2(e) into Q so softmax exponentials are pure exp2
  const float qscale = 0.08838834764831845f * LOG2E;

  // ---- preload Q^T as 4 B-fragments (K-dim = dk, N = query) ----
  // B layout (32x16 bf16): lane(g,n) holds column n, K = 16*g + e (e = 0..15 contiguous)
  v16bf qf[4];
  #pragma unroll
  for (int c = 0; c < 4; ++c) {
    const float* qp = Q + (qrow0 + qbase + n) * DKd + c * 32 + g * 16;
    float t[16];
    ((float4*)t)[0] = ((const float4*)qp)[0];
    ((float4*)t)[1] = ((const float4*)qp)[1];
    ((float4*)t)[2] = ((const float4*)qp)[2];
    ((float4*)t)[3] = ((const float4*)qp)[3];
    AFrag af;
    #pragma unroll
    for (int e = 0; e < 16; ++e) af.h[e] = f2bf(t[e] * qscale);
    qf[c] = af.v;
  }

  // O^T accumulator: 8 tiles of 16(dv) x 16(query), f32 D layout
  v8f oacc[8];
  const v8f vzero = {0.f,0.f,0.f,0.f,0.f,0.f,0.f,0.f};
  #pragma unroll
  for (int t = 0; t < 8; ++t) oacc[t] = vzero;
  float m = -1e30f, l = 0.0f;

  // K staging: 256 threads cover 64 rows x 128 cols; 32 floats each
  const int ksrow  = tid >> 2;         // key row 0..63
  const int kscol  = (tid & 3) * 32;   // dk col base {0,32,64,96}
  // V staging (transposed, dword-packed: two keys per ds_store_b32)
  const int vkpair = tid >> 3;         // key pair 0..31 -> keys {2p, 2p+1}
  const int vdv0   = (tid & 7) * 16;   // dv base 0..112

  for (int kc = 0; kc < SKd / KCHUNK; ++kc) {
    // ---- cooperative staging: K -> LDS(bf16 row-major), V -> LDS(bf16 transposed) ----
    {
      const float* kp = K + (krow0 + kc * KCHUNK + ksrow) * DKd + kscol;
      float tk[32];
      #pragma unroll
      for (int i = 0; i < 8; ++i) ((float4*)tk)[i] = ((const float4*)kp)[i];
      AFrag af0, af1;
      #pragma unroll
      for (int e = 0; e < 16; ++e) { af0.h[e] = f2bf(tk[e]); af1.h[e] = f2bf(tk[16 + e]); }
      uint4* dst = (uint4*)&lds_k[ksrow * KLDS_STRIDE + kscol];
      dst[0] = af0.q[0]; dst[1] = af0.q[1];
      dst[2] = af1.q[0]; dst[3] = af1.q[1];

      const float* vp0 = V + (krow0 + kc * KCHUNK + 2 * vkpair) * DVd + vdv0;
      const float* vp1 = vp0 + DVd;
      float ta[16], tb[16];
      #pragma unroll
      for (int i = 0; i < 4; ++i) { ((float4*)ta)[i] = ((const float4*)vp0)[i];
                                    ((float4*)tb)[i] = ((const float4*)vp1)[i]; }
      unsigned* vdst = (unsigned*)&lds_vt[0];
      #pragma unroll
      for (int e = 0; e < 16; ++e)
        vdst[((vdv0 + e) * VLDS_STRIDE + 2 * vkpair) >> 1] = pk2(ta[e], tb[e]);

      if (kc + 1 < SKd / KCHUNK) {       // -> global_prefetch_b8
        __builtin_prefetch(kp + KCHUNK * DKd, 0, 3);
        __builtin_prefetch(vp0 + KCHUNK * DVd, 0, 3);
      }
    }
    __syncthreads();

    // ---- S^T = K * Q^T : four 16x16 D tiles (keys 16i..16i+15 on M, queries on N) ----
    // A layout (16x32 bf16): lane(g, m=n) row m; elems 0..15: k=16g+j (contiguous, same order as the Q^T B frags)
    v8f s[4];
    #pragma unroll
    for (int i = 0; i < 4; ++i) s[i] = vzero;
    #pragma unroll
    for (int c = 0; c < 4; ++c) {
      #pragma unroll
      for (int i = 0; i < 4; ++i) {
        AFrag a;
        const unsigned short* kb = &lds_k[(16 * i + n) * KLDS_STRIDE + c * 32 + g * 16];
        a.q[0] = *(const uint4*)kb;
        a.q[1] = *(const uint4*)(kb + 8);
        s[i] = __builtin_amdgcn_wmma_f32_16x16x32_bf16(false, a.v, false, qf[c], (short)0, s[i], false, false);
      }
    }

    // ---- online softmax over the 64 keys of this chunk (per query column) ----
    float tmax = -1e30f;
    #pragma unroll
    for (int i = 0; i < 4; ++i)
      #pragma unroll
      for (int r = 0; r < 8; ++r) tmax = fmaxf(tmax, s[i][r]);
    tmax = fmaxf(tmax, swz16f(tmax));            // combine the two half-wave key groups
    const float mnew = fmaxf(m, tmax);
    const float corr = __builtin_amdgcn_exp2f(m - mnew);
    float p[4][8];
    float rsum = 0.0f;
    #pragma unroll
    for (int i = 0; i < 4; ++i)
      #pragma unroll
      for (int r = 0; r < 8; ++r) {
        p[i][r] = __builtin_amdgcn_exp2f(s[i][r] - mnew);
        rsum += p[i][r];
      }
    rsum += swz16f(rsum);
    l = l * corr + rsum;
    m = mnew;
    #pragma unroll
    for (int t = 0; t < 8; ++t)
      #pragma unroll
      for (int r = 0; r < 8; ++r) oacc[t][r] *= corr;

    // ---- convert P^T (D layout, f32) -> two B fragments (32x16 bf16 each) ----
    // B lane(g,n) needs K = 16g+2v,2v+1 in VGPR v. D tile VGPR r holds key r+8g.
    // -> per-lane bf16 pack + one half-wave swap per VGPR, no LDS traffic.
    AFrag pb[2];
    #pragma unroll
    for (int f = 0; f < 2; ++f) {
      unsigned q0[4], q1[4];
      #pragma unroll
      for (int j = 0; j < 4; ++j) {
        q0[j] = pk2(p[2*f][2*j],   p[2*f][2*j+1]);
        q1[j] = pk2(p[2*f+1][2*j], p[2*f+1][2*j+1]);
      }
      #pragma unroll
      for (int jv = 0; jv < 4; ++jv) {
        const unsigned sw1 = swz16u(q1[jv]);
        const unsigned sw0 = swz16u(q0[jv]);
        pb[f].u[jv]     = (g == 0) ? q0[jv] : sw1;
        pb[f].u[jv + 4] = (g == 0) ? sw0    : q1[jv];
      }
    }

    // ---- O^T += V^T * P^T : 8 dv tiles x 2 key fragments ----
    #pragma unroll
    for (int t = 0; t < 8; ++t) {
      #pragma unroll
      for (int f = 0; f < 2; ++f) {
        AFrag av;
        const unsigned short* vb = &lds_vt[(t * 16 + n) * VLDS_STRIDE + f * 32 + g * 16];
        av.q[0] = *(const uint4*)vb;
        av.q[1] = *(const uint4*)(vb + 8);
        oacc[t] = __builtin_amdgcn_wmma_f32_16x16x32_bf16(false, av.v, false, pb[f].v, (short)0, oacc[t], false, false);
      }
    }
    __syncthreads();
  }

  // ---- epilogue: O[q][dv] = O^T[dv][q] / l ----
  const float invl = 1.0f / l;
  float* op = O + (qrow0 + qbase + n) * DVd;
  #pragma unroll
  for (int t = 0; t < 8; ++t) {
    float4 w0, w1;
    w0.x = oacc[t][0] * invl; w0.y = oacc[t][1] * invl;
    w0.z = oacc[t][2] * invl; w0.w = oacc[t][3] * invl;
    w1.x = oacc[t][4] * invl; w1.y = oacc[t][5] * invl;
    w1.z = oacc[t][6] * invl; w1.w = oacc[t][7] * invl;
    *(float4*)&op[t * 16 + g * 8]     = w0;
    *(float4*)&op[t * 16 + g * 8 + 4] = w1;
  }
}

extern "C" void kernel_launch(void* const* d_in, const int* in_sizes, int n_in,
                              void* d_out, int out_size, void* d_ws, size_t ws_size,
                              hipStream_t stream) {
  (void)in_sizes; (void)n_in; (void)d_ws; (void)ws_size; (void)out_size;
  const float* Q = (const float*)d_in[0];
  const float* K = (const float*)d_in[1];
  const float* V = (const float*)d_in[2];
  float* O = (float*)d_out;
  // 64 (b,h) pairs * 16 query tiles of 128 rows = 1024 blocks; 256 threads = 8 waves
  dim3 grid(64 * 16), block(256);
  hipLaunchKernelGGL(fa_fwd_bf16_wmma, grid, block, 0, stream, Q, K, V, O);
}
